// MipNerfModel_33191507264172
// MI455X (gfx1250) — compile-verified, matrix operand order checked by reading, not run on hardware
//
#include <hip/hip_runtime.h>
#include <hip/hip_bf16.h>
#include <math.h>

typedef __attribute__((ext_vector_type(16))) _Float16 v16h;
typedef __attribute__((ext_vector_type(8)))  _Float16 v8h;
typedef __attribute__((ext_vector_type(8)))  float    v8f;

#define NUM_B   512
#define NSAMP   128
#define NTOT    (NUM_B * NSAMP)
#define NBIN    (NSAMP + 1)
#define LD_ACT  320
#define DENS_BIAS   (-1.0f)
#define RGB_PAD_F   0.001f
#define RESAMPLE_PD 0.01f
#define PI_F 3.14159265358979f

// ---------------------------------------------------------------------------
// Weight conversion: fp32 (K x N, row major) -> f16 transposed padded (N x Kpad)
// ---------------------------------------------------------------------------
__global__ void convert_wT(const float* __restrict__ src, int Kreal, int N,
                           _Float16* __restrict__ dst, int Kpad)
{
    int idx = blockIdx.x * blockDim.x + threadIdx.x;
    if (idx >= N * Kpad) return;
    int n = idx / Kpad;
    int k = idx - n * Kpad;
    float v = (k < Kreal) ? src[(size_t)k * N + n] : 0.0f;
    dst[idx] = (_Float16)v;
}

__global__ void zero_f32(float* __restrict__ p, int n)
{
    int i = blockIdx.x * blockDim.x + threadIdx.x;
    if (i < n) p[i] = 0.0f;
}

// ---------------------------------------------------------------------------
// WMMA GEMM, 32x32 register-blocked: one wave computes a 32x32 output tile
// using 2 A fragments x 2 B fragments -> 4 accumulators (4 WMMA per K-step).
// B supplied transposed: BT[n*ldbt + k].
// A fragment (16x32 f16): lane L (m = L&15): elem e -> K = 16*(e/8) + 8*(L>>4) + e%8
// B fragment mirrors with n = L&15.  D: vgpr r, lane L -> m = r + 8*(L>>4), n = L&15.
// ---------------------------------------------------------------------------
__global__ __launch_bounds__(256)
void gemm_wmma(const _Float16* __restrict__ A, int lda,
               const _Float16* __restrict__ BT, int ldbt,
               const float* __restrict__ bias,
               _Float16* __restrict__ C, int ldc,
               int tiles_n, int total_tiles, int K, int relu)
{
    int wave = threadIdx.x >> 5;
    int lane = threadIdx.x & 31;
    int tile = blockIdx.x * 8 + wave;          // wave-uniform
    if (tile >= total_tiles) return;           // whole wave exits together
    int tm = tile / tiles_n;
    int tn = tile - tm * tiles_n;
    int l16  = lane & 15;
    int half = lane >> 4;

    const _Float16* arow0 = A  + (size_t)(tm * 32 + l16)      * lda  + half * 8;
    const _Float16* arow1 = A  + (size_t)(tm * 32 + 16 + l16) * lda  + half * 8;
    const _Float16* brow0 = BT + (size_t)(tn * 32 + l16)      * ldbt + half * 8;
    const _Float16* brow1 = BT + (size_t)(tn * 32 + 16 + l16) * ldbt + half * 8;

    v8f acc00, acc01, acc10, acc11;
#pragma unroll
    for (int r = 0; r < 8; ++r) { acc00[r] = 0.0f; acc01[r] = 0.0f; acc10[r] = 0.0f; acc11[r] = 0.0f; }

    for (int k0 = 0; k0 < K; k0 += 32) {
        union { v16h v; v8h h[2]; } a0, a1, b0, b1;
        a0.h[0] = *(const v8h*)(arow0 + k0);
        a0.h[1] = *(const v8h*)(arow0 + k0 + 16);
        a1.h[0] = *(const v8h*)(arow1 + k0);
        a1.h[1] = *(const v8h*)(arow1 + k0 + 16);
        b0.h[0] = *(const v8h*)(brow0 + k0);
        b0.h[1] = *(const v8h*)(brow0 + k0 + 16);
        b1.h[0] = *(const v8h*)(brow1 + k0);
        b1.h[1] = *(const v8h*)(brow1 + k0 + 16);
        acc00 = __builtin_amdgcn_wmma_f32_16x16x32_f16(false, a0.v, false, b0.v,
                                                       (short)0, acc00, false, false);
        acc01 = __builtin_amdgcn_wmma_f32_16x16x32_f16(false, a0.v, false, b1.v,
                                                       (short)0, acc01, false, false);
        acc10 = __builtin_amdgcn_wmma_f32_16x16x32_f16(false, a1.v, false, b0.v,
                                                       (short)0, acc10, false, false);
        acc11 = __builtin_amdgcn_wmma_f32_16x16x32_f16(false, a1.v, false, b1.v,
                                                       (short)0, acc11, false, false);
    }

    int n0 = tn * 32 + l16;
    int n1 = n0 + 16;
    float bv0 = bias ? bias[n0] : 0.0f;
    float bv1 = bias ? bias[n1] : 0.0f;
    int r0 = tm * 32 + half * 8;       // rows for acc0x
    int r1 = r0 + 16;                  // rows for acc1x
#pragma unroll
    for (int r = 0; r < 8; ++r) {
        float v00 = acc00[r] + bv0;
        float v01 = acc01[r] + bv1;
        float v10 = acc10[r] + bv0;
        float v11 = acc11[r] + bv1;
        if (relu) {
            v00 = fmaxf(v00, 0.0f); v01 = fmaxf(v01, 0.0f);
            v10 = fmaxf(v10, 0.0f); v11 = fmaxf(v11, 0.0f);
        }
        C[(size_t)(r0 + r) * ldc + n0] = (_Float16)v00;
        C[(size_t)(r0 + r) * ldc + n1] = (_Float16)v01;
        C[(size_t)(r1 + r) * ldc + n0] = (_Float16)v10;
        C[(size_t)(r1 + r) * ldc + n1] = (_Float16)v11;
    }
}

// ---------------------------------------------------------------------------
// Ray setup: aa2matrix, object-frame transform, ray-box intersect, view enc
// ---------------------------------------------------------------------------
__global__ void ray_setup(const float* __restrict__ origins, const float* __restrict__ directions,
                          const float* __restrict__ viewdirs, const int* __restrict__ ts,
                          const float* __restrict__ init, const float* __restrict__ ext,
                          float* __restrict__ inter, float* __restrict__ orig_s,
                          float* __restrict__ dir_s, float* __restrict__ dnorm,
                          _Float16* __restrict__ venc)
{
    int b = blockIdx.x * blockDim.x + threadIdx.x;
    if (b >= NUM_B) return;
    float o[3], d[3];
#pragma unroll
    for (int c = 0; c < 3; ++c) { o[c] = origins[b*3+c]; d[c] = directions[b*3+c]; }
    int t = ts[b];
    float hit[2];
    float osum[3] = {0,0,0}, dsum[3] = {0,0,0};

    for (int i = 0; i < 2; ++i) {
        const float* p = init + ((size_t)t * 2 + i) * 6;
        float ax = p[3], ay = p[4], az = p[5];
        float th = sqrtf(ax*ax + ay*ay + az*az + 1e-12f);
        float k[3] = { ax/th, ay/th, az/th };
        float s  = __sinf(th);
        float c1 = 1.0f - __cosf(th);
        float Km[3][3] = { {0.f, -k[2],  k[1]},
                           {k[2], 0.f,  -k[0]},
                           {-k[1], k[0], 0.f} };
        float R[3][3];
#pragma unroll
        for (int a = 0; a < 3; ++a)
#pragma unroll
            for (int bb = 0; bb < 3; ++bb) {
                float dl = (a == bb) ? 1.0f : 0.0f;
                R[a][bb] = dl + s * Km[a][bb] + c1 * (k[a] * k[bb] - dl);
            }
        float rel[3] = { o[0]-p[0], o[1]-p[1], o[2]-p[2] };
        float oo[3], dd[3];
#pragma unroll
        for (int a = 0; a < 3; ++a) {      // Rt @ v  ==  sum_j R[j][a] v[j]
            oo[a] = R[0][a]*rel[0] + R[1][a]*rel[1] + R[2][a]*rel[2];
            dd[a] = R[0][a]*d[0]   + R[1][a]*d[1]   + R[2][a]*d[2];
        }
        float zi = -3.4e38f, zo = 3.4e38f;
#pragma unroll
        for (int c = 0; c < 3; ++c) {
            float den = (fabsf(dd[c]) < 1e-9f) ? 1e-9f : dd[c];
            float inv = 1.0f / den;
            float e = ext[i*3 + c];
            float t1 = (-e - oo[c]) * inv;
            float t2 = ( e - oo[c]) * inv;
            zi = fmaxf(zi, fminf(t1, t2));
            zo = fminf(zo, fmaxf(t1, t2));
        }
        float h = (zo > zi && zo > 0.0f) ? 1.0f : 0.0f;
        hit[i] = h;
        inter[b*2 + i] = h;
#pragma unroll
        for (int c = 0; c < 3; ++c) { osum[c] += h * oo[c]; dsum[c] += h * dd[c]; }
    }
    float bg = (hit[0] + hit[1] == 0.0f) ? 1.0f : 0.0f;
    float dn2 = 0.0f;
#pragma unroll
    for (int c = 0; c < 3; ++c) {
        float ov = osum[c] + bg * o[c];
        float dv = dsum[c] + bg * d[c];
        orig_s[b*3+c] = ov;
        dir_s [b*3+c] = dv;
        dn2 += dv * dv;
    }
    dnorm[b] = sqrtf(dn2);

    // viewdirs pos_enc: [v(3), sin(2^j v) j=0..3 (12), cos(2^j v) (12)] pad->32
    float v[3];
#pragma unroll
    for (int c = 0; c < 3; ++c) v[c] = viewdirs[b*3+c];
    _Float16* ve = venc + (size_t)b * 32;
#pragma unroll
    for (int c = 0; c < 3; ++c) ve[c] = (_Float16)v[c];
    for (int j = 0; j < 4; ++j) {
        float sc = (float)(1 << j);
#pragma unroll
        for (int c = 0; c < 3; ++c) {
            float y = v[c] * sc;
            ve[3  + j*3 + c] = (_Float16)__sinf(y);
            ve[15 + j*3 + c] = (_Float16)__cosf(y);
        }
    }
    for (int kk = 27; kk < 32; ++kk) ve[kk] = (_Float16)0.0f;
}

// ---------------------------------------------------------------------------
// t_vals (level 0), cone casting
// ---------------------------------------------------------------------------
__global__ void tvals_init(const float* __restrict__ nearp, const float* __restrict__ farp,
                           float* __restrict__ tv)
{
    int idx = blockIdx.x * blockDim.x + threadIdx.x;
    if (idx >= NUM_B * NBIN) return;
    int b = idx / NBIN;
    int j = idx - b * NBIN;
    float t = (float)j * (1.0f / (float)NSAMP);
    tv[idx] = nearp[b] + (farp[b] - nearp[b]) * t;
}

__global__ void cast_cone_kernel(const float* __restrict__ tv,
                                 const float* __restrict__ orig_s, const float* __restrict__ dir_s,
                                 const float* __restrict__ radii,
                                 float* __restrict__ meanb, float* __restrict__ varb)
{
    int s = blockIdx.x * blockDim.x + threadIdx.x;
    if (s >= NTOT) return;
    int b = s / NSAMP;
    int j = s - b * NSAMP;
    float t0 = tv[b*NBIN + j];
    float t1 = tv[b*NBIN + j + 1];
    float mu = 0.5f * (t0 + t1);
    float hw = 0.5f * (t1 - t0);
    float mu2 = mu*mu, hw2 = hw*hw, hw4 = hw2*hw2;
    float denom = 3.0f*mu2 + hw2;
    float t_mean = mu + 2.0f*mu*hw2 / denom;
    float t_var  = hw2/3.0f - (4.0f/15.0f) * (hw4 * (12.0f*mu2 - hw2) / (denom*denom));
    float r = radii[b];
    float r_var = r*r * (mu2*0.25f + (5.0f/12.0f)*hw2 - (4.0f/15.0f)*hw4/denom);
    float dv[3], d2[3];
    float dmag2 = 0.0f;
#pragma unroll
    for (int c = 0; c < 3; ++c) { dv[c] = dir_s[b*3+c]; d2[c] = dv[c]*dv[c]; dmag2 += d2[c]; }
    dmag2 = fmaxf(1e-10f, dmag2);
#pragma unroll
    for (int c = 0; c < 3; ++c) {
        meanb[s*3+c] = orig_s[b*3+c] + dv[c] * t_mean;
        varb [s*3+c] = t_var * d2[c] + r_var * (1.0f - d2[c]/dmag2);
    }
}

// ---------------------------------------------------------------------------
// Weighted IPE encoding -> f16 [NTOT, 64]
// ---------------------------------------------------------------------------
__device__ __forceinline__ float obj_mask(const float* inter, int b, int inst)
{
    return (inst < 2) ? inter[b*2 + inst] : (1.0f - inter[b*2] - inter[b*2+1]);
}

__global__ void encode_kernel(const float* __restrict__ meanb, const float* __restrict__ varb,
                              const float* __restrict__ inter, const float* __restrict__ alphap,
                              int inst, _Float16* __restrict__ enc)
{
    int s = blockIdx.x * blockDim.x + threadIdx.x;
    if (s >= NTOT) return;
    int b = s / NSAMP;
    float m = obj_mask(inter, b, inst);
    float alpha = alphap[0];
    float mx[3], vx[3];
#pragma unroll
    for (int c = 0; c < 3; ++c) { mx[c] = m*meanb[s*3+c]; vx[c] = m*varb[s*3+c]; }
    _Float16* o = enc + (size_t)s * 64;
    for (int j = 0; j < 10; ++j) {
        float sc = (float)(1 << j);
        float a = fminf(fmaxf(alpha - (float)j, 0.0f), 1.0f);
        float w = 0.5f * (1.0f - __cosf(PI_F * a));
#pragma unroll
        for (int c = 0; c < 3; ++c) {
            float y  = mx[c] * sc;
            float yv = vx[c] * sc * sc;
            float e  = __expf(-0.5f * yv) * w;
            o[j*3 + c]      = (_Float16)(e * __sinf(y));
            o[30 + j*3 + c] = (_Float16)(e * __cosf(y));
        }
    }
#pragma unroll
    for (int k = 60; k < 64; ++k) o[k] = (_Float16)0.0f;
}

// ---------------------------------------------------------------------------
// Concat helpers: copy ENC (64) / viewdir enc (32, per-ray) into ACT tail cols
// ---------------------------------------------------------------------------
__global__ void append_enc(const _Float16* __restrict__ enc, _Float16* __restrict__ act, int W)
{
    int idx = blockIdx.x * blockDim.x + threadIdx.x;
    if (idx >= NTOT * 8) return;
    int s = idx >> 3;
    int c = idx & 7;
    *(v8h*)(act + (size_t)s * LD_ACT + W + c * 8) = *(const v8h*)(enc + (size_t)s * 64 + c * 8);
}

__global__ void append_view(const _Float16* __restrict__ venc, _Float16* __restrict__ act, int W)
{
    int idx = blockIdx.x * blockDim.x + threadIdx.x;
    if (idx >= NTOT * 4) return;
    int s = idx >> 2;
    int c = idx & 3;
    int b = s / NSAMP;
    *(v8h*)(act + (size_t)s * LD_ACT + W + c * 8) = *(const v8h*)(venc + (size_t)b * 32 + c * 8);
}

// ---------------------------------------------------------------------------
// Heads: density (N=1) and rgb (N=3) GEMVs with masked accumulation
// ---------------------------------------------------------------------------
__global__ void density_head(const _Float16* __restrict__ X, const float* __restrict__ w,
                             const float* __restrict__ bias, const float* __restrict__ inter,
                             int inst, float* __restrict__ raw_den, int W)
{
    int s = blockIdx.x * blockDim.x + threadIdx.x;
    if (s >= NTOT) return;
    const _Float16* x = X + (size_t)s * LD_ACT;
    float acc = bias[0];
    for (int k = 0; k < W; k += 8) {
        v8h xv = *(const v8h*)(x + k);
#pragma unroll
        for (int e = 0; e < 8; ++e) acc += (float)xv[e] * w[k + e];
    }
    raw_den[s] += obj_mask(inter, s / NSAMP, inst) * acc;
}

__global__ void rgb_head(const _Float16* __restrict__ X, const float* __restrict__ w,
                         const float* __restrict__ bias, const float* __restrict__ inter,
                         int inst, float* __restrict__ raw_rgb)
{
    int s = blockIdx.x * blockDim.x + threadIdx.x;
    if (s >= NTOT) return;
    const _Float16* x = X + (size_t)s * LD_ACT;
    float a0 = bias[0], a1 = bias[1], a2 = bias[2];
    for (int k = 0; k < 128; k += 8) {
        v8h xv = *(const v8h*)(x + k);
#pragma unroll
        for (int e = 0; e < 8; ++e) {
            float f = (float)xv[e];
            a0 += f * w[(k + e)*3 + 0];
            a1 += f * w[(k + e)*3 + 1];
            a2 += f * w[(k + e)*3 + 2];
        }
    }
    float m = obj_mask(inter, s / NSAMP, inst);
    raw_rgb[s*3 + 0] += m * a0;
    raw_rgb[s*3 + 1] += m * a1;
    raw_rgb[s*3 + 2] += m * a2;
}

// ---------------------------------------------------------------------------
// Volumetric rendering (serial per ray) — emits comp_rgb and weights
// ---------------------------------------------------------------------------
__global__ void render_kernel(const float* __restrict__ raw_rgb, const float* __restrict__ raw_den,
                              const float* __restrict__ tv, const float* __restrict__ dnorm,
                              float* __restrict__ wbuf, float* __restrict__ out)
{
    int b = blockIdx.x * blockDim.x + threadIdx.x;
    if (b >= NUM_B) return;
    float dn = dnorm[b];
    float T = 1.0f;
    float acc[3] = {0,0,0};
    for (int j = 0; j < NSAMP; ++j) {
        int s = b * NSAMP + j;
        float delta = (tv[b*NBIN + j + 1] - tv[b*NBIN + j]) * dn;
        float rd = raw_den[s] + DENS_BIAS;
        float dens = (rd > 20.0f) ? rd : log1pf(__expf(rd));      // softplus
        float dd = dens * delta;
        float al = 1.0f - __expf(-dd);
        float w = al * T;
        T *= __expf(-dd);
        wbuf[s] = w;
#pragma unroll
        for (int c = 0; c < 3; ++c) {
            float x = raw_rgb[s*3 + c];
            float sg = 1.0f / (1.0f + __expf(-x));
            acc[c] += w * (sg * (1.0f + 2.0f * RGB_PAD_F) - RGB_PAD_F);
        }
    }
#pragma unroll
    for (int c = 0; c < 3; ++c) out[b*3 + c] = acc[c];
}

// ---------------------------------------------------------------------------
// Inverse-CDF resampling (one block per ray)
// ---------------------------------------------------------------------------
__global__ __launch_bounds__(256)
void resample_kernel(const float* __restrict__ tv_in, const float* __restrict__ wbuf,
                     float* __restrict__ tv_out)
{
    __shared__ float bins[NBIN];
    __shared__ float cdf[NBIN];
    __shared__ float wnew[NSAMP];
    int b = blockIdx.x;
    int t = threadIdx.x;
    if (t < NBIN) bins[t] = tv_in[b*NBIN + t];
    if (t < NSAMP) {
        int im1 = (t - 1 < 0) ? 0 : t - 1;
        int ip1 = (t + 1 > NSAMP - 1) ? NSAMP - 1 : t + 1;
        float w0 = wbuf[b*NSAMP + im1];
        float w1 = wbuf[b*NSAMP + t];
        float w2 = wbuf[b*NSAMP + ip1];
        wnew[t] = 0.5f * (fmaxf(w0, w1) + fmaxf(w1, w2)) + RESAMPLE_PD;
    }
    __syncthreads();
    if (t == 0) {
        float s = 0.0f;
        for (int i = 0; i < NSAMP; ++i) s += wnew[i];
        float pad = fmaxf(0.0f, 1e-5f - s);
        float wsum = s + pad;
        float padper = pad / (float)NSAMP;
        float c = 0.0f;
        cdf[0] = 0.0f;
        for (int i = 1; i <= NSAMP - 1; ++i) {
            c += (wnew[i-1] + padper) / wsum;
            cdf[i] = fminf(1.0f, c);
        }
        cdf[NSAMP] = 1.0f;
    }
    __syncthreads();
    if (t < NBIN) {
        float u = (float)t * (1.0f - 1.1920929e-7f) / (float)NSAMP;
        int i0 = 0;
        for (int i = 1; i <= NSAMP; ++i)
            if (u >= cdf[i]) i0 = i;
        int i1 = (i0 + 1 > NSAMP) ? NSAMP : i0 + 1;
        float c0 = cdf[i0], c1 = cdf[i1];
        float den = c1 - c0;
        float tt = (den != 0.0f) ? (u - c0) / den : 0.0f;
        tt = fminf(fmaxf(tt, 0.0f), 1.0f);
        tv_out[b*NBIN + t] = bins[i0] + tt * (bins[i1] - bins[i0]);
    }
}

// ---------------------------------------------------------------------------
// Host orchestration
// ---------------------------------------------------------------------------
struct MlpHost {
    const float* lw[8]; const float* lb[8];
    const float *dw, *db, *bw, *bb, *cw, *cb, *rw, *rb;
    int W;
};

struct MlpDev {
    _Float16 *w0, *w1, *w2, *w3, *w4, *w5, *w6, *w7, *bott, *cond;
};

static void fill_mlp(MlpHost& m, void* const* d_in, int base, int W)
{
    m.bb = (const float*)d_in[base + 0];
    m.bw = (const float*)d_in[base + 1];
    m.cb = (const float*)d_in[base + 2];
    m.cw = (const float*)d_in[base + 3];
    m.db = (const float*)d_in[base + 4];
    m.dw = (const float*)d_in[base + 5];
    for (int i = 0; i < 8; ++i) {
        m.lb[i] = (const float*)d_in[base + 6 + 2*i];
        m.lw[i] = (const float*)d_in[base + 7 + 2*i];
    }
    m.rb = (const float*)d_in[base + 22];
    m.rw = (const float*)d_in[base + 23];
    m.W = W;
}

static void* ws_alloc(char* base, size_t* off, size_t bytes)
{
    *off = (*off + 255) & ~(size_t)255;
    void* p = base + *off;
    *off += bytes;
    return p;
}

static _Float16* conv_w(hipStream_t st, char* ws, size_t* off,
                        const float* src, int Kreal, int N, int Kpad)
{
    _Float16* dst = (_Float16*)ws_alloc(ws, off, (size_t)N * Kpad * sizeof(_Float16));
    int n = N * Kpad;
    convert_wT<<<(n + 255) / 256, 256, 0, st>>>(src, Kreal, N, dst, Kpad);
    return dst;
}

static void launch_gemm(hipStream_t st, const _Float16* A, int lda,
                        const _Float16* BT, int ldbt, const float* bias,
                        _Float16* C, int ldc, int N, int K, int relu)
{
    int tiles_m = NTOT / 32;
    int tiles_n = N / 32;
    int total = tiles_m * tiles_n;
    gemm_wmma<<<(total + 7) / 8, 256, 0, st>>>(A, lda, BT, ldbt, bias, C, ldc,
                                               tiles_n, total, K, relu);
}

extern "C" void kernel_launch(void* const* d_in, const int* in_sizes, int n_in,
                              void* d_out, int out_size, void* d_ws, size_t ws_size,
                              hipStream_t stream)
{
    (void)in_sizes; (void)n_in; (void)out_size; (void)ws_size;
    const float* origins    = (const float*)d_in[0];
    const float* directions = (const float*)d_in[1];
    const float* viewdirs   = (const float*)d_in[2];
    const float* radii      = (const float*)d_in[3];
    const float* nearp      = (const float*)d_in[4];
    const float* farp       = (const float*)d_in[5];
    const float* init       = (const float*)d_in[6];
    const float* ext        = (const float*)d_in[7];
    const float* alphap     = (const float*)d_in[8];
    const int*   ts         = (const int*)d_in[9];

    MlpHost hostp[3];
    fill_mlp(hostp[0], d_in, 34, 128);   // params_obj[0]
    fill_mlp(hostp[1], d_in, 58, 128);   // params_obj[1]
    fill_mlp(hostp[2], d_in, 10, 256);   // params_mlp (background)

    char* ws = (char*)d_ws;
    size_t off = 0;
    float*     interb  = (float*)ws_alloc(ws, &off, NUM_B * 2 * 4);
    float*     orig_s  = (float*)ws_alloc(ws, &off, NUM_B * 3 * 4);
    float*     dir_s   = (float*)ws_alloc(ws, &off, NUM_B * 3 * 4);
    float*     dnormb  = (float*)ws_alloc(ws, &off, NUM_B * 4);
    _Float16*  venc    = (_Float16*)ws_alloc(ws, &off, NUM_B * 32 * 2);
    float*     tv0     = (float*)ws_alloc(ws, &off, NUM_B * NBIN * 4);
    float*     tv1     = (float*)ws_alloc(ws, &off, NUM_B * NBIN * 4);
    float*     meanb   = (float*)ws_alloc(ws, &off, (size_t)NTOT * 3 * 4);
    float*     varb    = (float*)ws_alloc(ws, &off, (size_t)NTOT * 3 * 4);
    _Float16*  ENC     = (_Float16*)ws_alloc(ws, &off, (size_t)NTOT * 64 * 2);
    _Float16*  A0      = (_Float16*)ws_alloc(ws, &off, (size_t)NTOT * LD_ACT * 2);
    _Float16*  A1      = (_Float16*)ws_alloc(ws, &off, (size_t)NTOT * LD_ACT * 2);
    float*     raw_rgb = (float*)ws_alloc(ws, &off, (size_t)NTOT * 3 * 4);
    float*     raw_den = (float*)ws_alloc(ws, &off, (size_t)NTOT * 4);
    float*     wbuf    = (float*)ws_alloc(ws, &off, (size_t)NTOT * 4);

    // --- convert all MLP weights to transposed / K-padded f16 ---------------
    MlpDev devw[3];
    for (int i = 0; i < 3; ++i) {
        const MlpHost& H = hostp[i];
        int W = H.W;
        devw[i].w0   = conv_w(stream, ws, &off, H.lw[0], 60,     W,   64);
        devw[i].w1   = conv_w(stream, ws, &off, H.lw[1], W,      W,   W);
        devw[i].w2   = conv_w(stream, ws, &off, H.lw[2], W,      W,   W);
        devw[i].w3   = conv_w(stream, ws, &off, H.lw[3], W,      W,   W);
        devw[i].w4   = conv_w(stream, ws, &off, H.lw[4], W,      W,   W);
        devw[i].w5   = conv_w(stream, ws, &off, H.lw[5], W + 60, W,   W + 64);
        devw[i].w6   = conv_w(stream, ws, &off, H.lw[6], W,      W,   W);
        devw[i].w7   = conv_w(stream, ws, &off, H.lw[7], W,      W,   W);
        devw[i].bott = conv_w(stream, ws, &off, H.bw,    W,      W,   W);
        devw[i].cond = conv_w(stream, ws, &off, H.cw,    W + 27, 128, W + 32);
    }

    // --- ray setup ----------------------------------------------------------
    ray_setup<<<(NUM_B + 63) / 64, 64, 0, stream>>>(origins, directions, viewdirs, ts,
                                                    init, ext, interb, orig_s, dir_s,
                                                    dnormb, venc);

    float* out = (float*)d_out;

    for (int level = 0; level < 2; ++level) {
        float* tv = (level == 0) ? tv0 : tv1;
        if (level == 0) {
            tvals_init<<<(NUM_B * NBIN + 255) / 256, 256, 0, stream>>>(nearp, farp, tv0);
        } else {
            resample_kernel<<<NUM_B, 256, 0, stream>>>(tv0, wbuf, tv1);
        }
        cast_cone_kernel<<<(NTOT + 255) / 256, 256, 0, stream>>>(tv, orig_s, dir_s,
                                                                 radii, meanb, varb);
        zero_f32<<<(NTOT * 3 + 255) / 256, 256, 0, stream>>>(raw_rgb, NTOT * 3);
        zero_f32<<<(NTOT + 255) / 256, 256, 0, stream>>>(raw_den, NTOT);

        for (int inst = 0; inst < 3; ++inst) {
            const MlpHost& H = hostp[inst];
            const MlpDev&  D = devw[inst];
            int W = H.W;
            encode_kernel<<<(NTOT + 255) / 256, 256, 0, stream>>>(meanb, varb, interb,
                                                                  alphap, inst, ENC);
            launch_gemm(stream, ENC, 64,     D.w0, 64,     H.lb[0], A0, LD_ACT, W, 64,     1);
            launch_gemm(stream, A0,  LD_ACT, D.w1, W,      H.lb[1], A1, LD_ACT, W, W,      1);
            launch_gemm(stream, A1,  LD_ACT, D.w2, W,      H.lb[2], A0, LD_ACT, W, W,      1);
            launch_gemm(stream, A0,  LD_ACT, D.w3, W,      H.lb[3], A1, LD_ACT, W, W,      1);
            launch_gemm(stream, A1,  LD_ACT, D.w4, W,      H.lb[4], A0, LD_ACT, W, W,      1);
            append_enc<<<(NTOT * 8 + 255) / 256, 256, 0, stream>>>(ENC, A0, W);
            launch_gemm(stream, A0,  LD_ACT, D.w5, W + 64, H.lb[5], A1, LD_ACT, W, W + 64, 1);
            launch_gemm(stream, A1,  LD_ACT, D.w6, W,      H.lb[6], A0, LD_ACT, W, W,      1);
            launch_gemm(stream, A0,  LD_ACT, D.w7, W,      H.lb[7], A1, LD_ACT, W, W,      1);
            density_head<<<(NTOT + 255) / 256, 256, 0, stream>>>(A1, H.dw, H.db, interb,
                                                                 inst, raw_den, W);
            launch_gemm(stream, A1,  LD_ACT, D.bott, W,      H.bb, A0, LD_ACT, W,   W,      0);
            append_view<<<(NTOT * 4 + 255) / 256, 256, 0, stream>>>(venc, A0, W);
            launch_gemm(stream, A0,  LD_ACT, D.cond, W + 32, H.cb, A1, LD_ACT, 128, W + 32, 1);
            rgb_head<<<(NTOT + 255) / 256, 256, 0, stream>>>(A1, H.rw, H.rb, interb,
                                                             inst, raw_rgb);
        }

        render_kernel<<<(NUM_B + 63) / 64, 64, 0, stream>>>(raw_rgb, raw_den, tv, dnormb,
                                                            wbuf, out + level * NUM_B * 3);
    }
}
